// Drug3DStructuralEncoder_56899726737560
// MI455X (gfx1250) — matrix operand order repaired and checked by reading, not compiled, not run on hardware
//
#include <hip/hip_runtime.h>

// ---------------------------------------------------------------------------
// GAT (2-layer) for MI455X / gfx1250.
// GEMMs via v_wmma_f32_16x16x32_bf16 (wave32 WMMA); edge phase is L2-resident
// gather/scatter with float-key atomicMax for segment-max.
// ---------------------------------------------------------------------------

typedef __bf16 bf16_t;
typedef __attribute__((ext_vector_type(16))) __bf16 v16bf;
typedef __attribute__((ext_vector_type(8)))  __bf16 v8bf;
typedef __attribute__((ext_vector_type(4)))  __bf16 v4bf;
typedef __attribute__((ext_vector_type(8)))  float  v8f;

#define GAT_HEADS    8
#define GAT_HDIM     32
#define GAT_DM       256
#define GAT_NEGSLOPE 0.2f
#define GAT_EPS      1e-16f

static constexpr int APITCH = 40;  // padded row pitch in bf16 elems (32 data + 8 pad)

// ---------------------------------------------------------------------------
// C[N,M] = act(A[N,K] @ W[K,M] (+bias)) ; bf16 WMMA, f32 accumulate.
// block = 128 threads (4 waves). Block tile: 64 rows x 64 cols.
// wave w handles rows [w*16, w*16+16), all 4 column sub-tiles.
// Staging is branch-free: fixed 4-iteration loops (no tid-dependent trip
// count), row indices clamped (garbage rows never stored by the epilogue).
// ---------------------------------------------------------------------------
__global__ void gemm_wmma_bf16(const float* __restrict__ A,
                               const float* __restrict__ W,
                               const float* __restrict__ bias,
                               float* __restrict__ C,
                               int Nrows, int K, int M,
                               int use_bias, int use_relu)
{
    __shared__ __align__(16) bf16_t Alds[64 * APITCH];
    __shared__ __align__(16) bf16_t Blds[64 * APITCH];   // B^T: [col][k]

    const int tid     = threadIdx.x;
    const int lane    = tid & 31;
    const int wave    = tid >> 5;            // 0..3
    const int rowBase = blockIdx.x * 64;
    const int colBase = blockIdx.y * 64;

    v8f acc[4] = {};                         // 4 n-tiles of 16x16 f32

    for (int ks = 0; ks < K; ks += 32) {
        // --- stage A tile: 64 rows x 32 k. float4 loads, packed bf16x4 stores.
        #pragma unroll
        for (int j = 0; j < 4; ++j) {
            const int i = tid * 4 + j * 512;           // 4*128 threads*4 = 2048 elems
            const int r = i >> 5, k = i & 31;          // k multiple of 4
            int gr = rowBase + r;
            gr = (gr < Nrows) ? gr : (Nrows - 1);      // clamp: branch-free
            const float4 f = *(const float4*)(A + (size_t)gr * K + ks + k);
            v4bf b;
            b[0] = (bf16_t)f.x; b[1] = (bf16_t)f.y;
            b[2] = (bf16_t)f.z; b[3] = (bf16_t)f.w;
            *(v4bf*)(Alds + r * APITCH + k) = b;
        }
        // --- stage B^T tile: float4 loads along cols, transposed b16 stores.
        #pragma unroll
        for (int j = 0; j < 4; ++j) {
            const int i = tid * 4 + j * 512;
            const int k = i >> 6, c = i & 63;          // c multiple of 4
            const float4 f = *(const float4*)(W + (size_t)(ks + k) * M + colBase + c);
            Blds[(c + 0) * APITCH + k] = (bf16_t)f.x;
            Blds[(c + 1) * APITCH + k] = (bf16_t)f.y;
            Blds[(c + 2) * APITCH + k] = (bf16_t)f.z;
            Blds[(c + 3) * APITCH + k] = (bf16_t)f.w;
        }
        __syncthreads();

        // --- A fragment (16-bit A layout: lanes 0-15 rows, K {kb..kb+7, kb+16..kb+23})
        const int arow = wave * 16 + (lane & 15);
        const int kb   = (lane < 16) ? 0 : 8;
        v8bf alo = *(const v8bf*)(Alds + arow * APITCH + kb);
        v8bf ahi = *(const v8bf*)(Alds + arow * APITCH + kb + 16);
        v16bf afrag;
        #pragma unroll
        for (int j = 0; j < 8; ++j) { afrag[j] = alo[j]; afrag[j + 8] = ahi[j]; }

        // --- load ALL 4 B fragments first so DS loads overlap the WMMA chain
        v16bf bfr[4];
        #pragma unroll
        for (int nt = 0; nt < 4; ++nt) {
            const int bcol = nt * 16 + (lane & 15);
            const int kb2  = (lane < 16) ? 0 : 8;
            v8bf blo = *(const v8bf*)(Blds + bcol * APITCH + kb2);
            v8bf bhi = *(const v8bf*)(Blds + bcol * APITCH + kb2 + 16);
            #pragma unroll
            for (int j = 0; j < 8; ++j) { bfr[nt][j] = blo[j]; bfr[nt][j + 8] = bhi[j]; }
        }

        #pragma unroll
        for (int nt = 0; nt < 4; ++nt) {
            acc[nt] = __builtin_amdgcn_wmma_f32_16x16x32_bf16(
                false, afrag, false, bfr[nt], (short)0, acc[nt], false, false);
        }
        __syncthreads();
    }

    // --- epilogue (C layout: lanes 0-15 -> M=vgpr, lanes 16-31 -> M=8+vgpr; N=lane%16)
    #pragma unroll
    for (int nt = 0; nt < 4; ++nt) {
        const int col = colBase + nt * 16 + (lane & 15);
        #pragma unroll
        for (int i = 0; i < 8; ++i) {
            const int r  = wave * 16 + ((lane < 16) ? i : (i + 8));
            const int gr = rowBase + r;
            if (gr < Nrows) {
                float v = acc[nt][i];
                if (use_bias) v += bias[col];
                if (use_relu) v = fmaxf(v, 0.0f);
                C[(size_t)gr * M + col] = v;
            }
        }
    }
}

// ---------------------------------------------------------------------------
// Per-node attention scalars: a_src[n,h] = <h[n,h,:], att_s[h,:]> (and a_dst).
// One wave32 per node; lane = channel; shuffle reduction.
// ---------------------------------------------------------------------------
__global__ void attn_scores(const float* __restrict__ hfeat,
                            const float* __restrict__ att_s,
                            const float* __restrict__ att_d,
                            float* __restrict__ a_src,
                            float* __restrict__ a_dst, int N)
{
    const int warp = threadIdx.x >> 5;
    const int lane = threadIdx.x & 31;
    const int n = blockIdx.x * 8 + warp;
    if (n >= N) return;
    const float* row = hfeat + (size_t)n * GAT_DM;
    #pragma unroll
    for (int hd = 0; hd < GAT_HEADS; ++hd) {
        float v = row[hd * GAT_HDIM + lane];
        float s = v * att_s[hd * GAT_HDIM + lane];
        float d = v * att_d[hd * GAT_HDIM + lane];
        #pragma unroll
        for (int o = 16; o > 0; o >>= 1) {
            s += __shfl_xor(s, o, 32);
            d += __shfl_xor(d, o, 32);
        }
        if (lane == 0) {
            a_src[n * GAT_HEADS + hd] = s;
            a_dst[n * GAT_HEADS + hd] = d;
        }
    }
}

__global__ void fill_u32(unsigned* __restrict__ p, unsigned v, int n)
{
    int i = blockIdx.x * blockDim.x + threadIdx.x;
    if (i < n) p[i] = v;
}

// Order-preserving float->uint key (radix trick): signed-float order == uint order.
__device__ __forceinline__ unsigned f32_key(float f)
{
    unsigned u = __float_as_uint(f);
    return (u & 0x80000000u) ? ~u : (u | 0x80000000u);
}

// Pass 1: e = leakyrelu(a_src[src] + a_dst[dst]); atomicMax of keyed e into mkey[dst].
__global__ void edge_scores(const long long* __restrict__ ei,
                            const float* __restrict__ a_src,
                            const float* __restrict__ a_dst,
                            float* __restrict__ ebuf,
                            unsigned* __restrict__ mkey, int E)
{
    int idx = blockIdx.x * blockDim.x + threadIdx.x;
    if (idx >= E * GAT_HEADS) return;
    int e = idx >> 3, hd = idx & 7;
    int s = (int)ei[e];
    int d = (int)ei[E + e];
    float v = a_src[s * GAT_HEADS + hd] + a_dst[d * GAT_HEADS + hd];
    v = (v > 0.0f) ? v : GAT_NEGSLOPE * v;
    ebuf[idx] = v;
    atomicMax(&mkey[d * GAT_HEADS + hd], f32_key(v));
}

// Decode max keys in place to floats (non-finite -> 0, matching reference) and zero denom.
__global__ void decode_max(float* __restrict__ m, float* __restrict__ denom, int n)
{
    int i = blockIdx.x * blockDim.x + threadIdx.x;
    if (i >= n) return;
    unsigned k = __float_as_uint(m[i]);                    // raw key bits
    unsigned u = (k & 0x80000000u) ? (k & 0x7FFFFFFFu) : ~k;
    float f = __uint_as_float(u);
    if ((u & 0x7F800000u) == 0x7F800000u) f = 0.0f;        // inf/nan -> 0
    m[i] = f;
    denom[i] = 0.0f;
}

// Pass 2: ex = exp(e - m[dst]); atomicAdd denom[dst].
__global__ void edge_exp(const long long* __restrict__ ei,
                         float* __restrict__ ebuf,
                         const float* __restrict__ m,
                         float* __restrict__ denom, int E)
{
    int idx = blockIdx.x * blockDim.x + threadIdx.x;
    if (idx >= E * GAT_HEADS) return;
    int e = idx >> 3, hd = idx & 7;
    int d = (int)ei[E + e];
    float ex = __expf(ebuf[idx] - m[d * GAT_HEADS + hd]);
    ebuf[idx] = ex;
    atomicAdd(&denom[d * GAT_HEADS + hd], ex);
}

// Pass 3: per edge, agg[dst,:] += h[src,:] * alpha[head]. One 256-thread block/edge;
// h and agg rows are L2-resident (51 MB each vs 192 MB L2).
__global__ void edge_aggregate(const long long* __restrict__ ei,
                               const float* __restrict__ hfeat,
                               const float* __restrict__ ebuf,
                               const float* __restrict__ denom,
                               float* __restrict__ agg, int E)
{
    const int e = blockIdx.x;
    const int c = threadIdx.x;               // 0..255
    const int hd = c >> 5;
    const int s = (int)ei[e];
    const int d = (int)ei[E + e];
    const float alpha = ebuf[e * GAT_HEADS + hd] /
                        (denom[d * GAT_HEADS + hd] + GAT_EPS);
    atomicAdd(&agg[(size_t)d * GAT_DM + c], hfeat[(size_t)s * GAT_DM + c] * alpha);
}

__global__ void bias_relu(const float* __restrict__ agg,
                          const float* __restrict__ bias,
                          float* __restrict__ out, int n)
{
    int i = blockIdx.x * blockDim.x + threadIdx.x;
    if (i < n) {
        float v = agg[i] + bias[i & (GAT_DM - 1)];
        out[i] = fmaxf(v, 0.0f);
    }
}

// ---------------------------------------------------------------------------
static void run_gat_layer(const float* xin, const float* W,
                          const float* att_s, const float* att_d, const float* b,
                          const long long* ei, int N, int E,
                          float* hbuf, float* asrc, float* adst,
                          float* mbuf, float* denom, float* ebuf, float* agg,
                          float* out, hipStream_t stream)
{
    dim3 gg((unsigned)((N + 63) / 64), GAT_DM / 64);
    gemm_wmma_bf16<<<gg, 128, 0, stream>>>(xin, W, nullptr, hbuf,
                                           N, GAT_DM, GAT_DM, 0, 0);

    attn_scores<<<(N + 7) / 8, 256, 0, stream>>>(hbuf, att_s, att_d, asrc, adst, N);

    const int nh = N * GAT_HEADS;
    const int eh = E * GAT_HEADS;
    const int nd = N * GAT_DM;

    // key of -inf: bits(-inf)=0xFF800000 (negative) -> ~bits = 0x007FFFFF
    fill_u32<<<(nh + 255) / 256, 256, 0, stream>>>((unsigned*)mbuf, 0x007FFFFFu, nh);
    edge_scores<<<(eh + 255) / 256, 256, 0, stream>>>(ei, asrc, adst, ebuf,
                                                      (unsigned*)mbuf, E);
    decode_max<<<(nh + 255) / 256, 256, 0, stream>>>(mbuf, denom, nh);
    edge_exp<<<(eh + 255) / 256, 256, 0, stream>>>(ei, ebuf, mbuf, denom, E);
    fill_u32<<<(nd + 255) / 256, 256, 0, stream>>>((unsigned*)agg, 0u, nd);
    edge_aggregate<<<E, 256, 0, stream>>>(ei, hbuf, ebuf, denom, agg, E);
    bias_relu<<<(nd + 255) / 256, 256, 0, stream>>>(agg, b, out, nd);
}

extern "C" void kernel_launch(void* const* d_in, const int* in_sizes, int n_in,
                              void* d_out, int out_size, void* d_ws, size_t ws_size,
                              hipStream_t stream)
{
    (void)n_in; (void)out_size; (void)ws_size;

    const float*     x   = (const float*)d_in[0];
    const long long* ei  = (const long long*)d_in[1];
    const float*     Wp  = (const float*)d_in[2];
    const float*     bp  = (const float*)d_in[3];
    const float*     W1  = (const float*)d_in[4];
    const float*     as1 = (const float*)d_in[5];
    const float*     ad1 = (const float*)d_in[6];
    const float*     b1  = (const float*)d_in[7];
    const float*     W2  = (const float*)d_in[8];
    const float*     as2 = (const float*)d_in[9];
    const float*     ad2 = (const float*)d_in[10];
    const float*     b2  = (const float*)d_in[11];

    const int IN_DIM = 128;
    const int N = in_sizes[0] / IN_DIM;
    const int E = in_sizes[1] / 2;

    // workspace carve-up (floats)
    float* ws = (float*)d_ws;
    size_t o = 0;
    float* xproj = ws + o; o += (size_t)N * GAT_DM;
    float* hbuf  = ws + o; o += (size_t)N * GAT_DM;
    float* h1    = ws + o; o += (size_t)N * GAT_DM;
    float* agg   = ws + o; o += (size_t)N * GAT_DM;
    float* asrc  = ws + o; o += (size_t)N * GAT_HEADS;
    float* adst  = ws + o; o += (size_t)N * GAT_HEADS;
    float* mbuf  = ws + o; o += (size_t)N * GAT_HEADS;
    float* denom = ws + o; o += (size_t)N * GAT_HEADS;
    float* ebuf  = ws + o; o += (size_t)E * GAT_HEADS;

    // x_proj = relu(x @ Wp + bp)
    dim3 g0((unsigned)((N + 63) / 64), GAT_DM / 64);
    gemm_wmma_bf16<<<g0, 128, 0, stream>>>(x, Wp, bp, xproj, N, IN_DIM, GAT_DM, 1, 1);

    // layer 1 -> h1 ; layer 2 -> d_out
    run_gat_layer(xproj, W1, as1, ad1, b1, ei, N, E,
                  hbuf, asrc, adst, mbuf, denom, ebuf, agg, h1, stream);
    run_gat_layer(h1, W2, as2, ad2, b2, ei, N, E,
                  hbuf, asrc, adst, mbuf, denom, ebuf, agg, (float*)d_out, stream);
}
